// PiecewiseShared_48430051229714
// MI455X (gfx1250) — matrix operand order repaired and checked by reading, not compiled, not run on hardware
//
#include <hip/hip_runtime.h>

// ---------------------------------------------------------------------------
// PiecewiseShared on MI455X (gfx1250, wave32, WMMA).
//
// out[b,o,m] = sum_{c,j} basis_j(t[b,c,m]) * w[o, c, 3*s[b,c,m] + j]
//
// Densified over the 97-long weight axis p (padded to 128):
//   out[(b,m), o] = sum_c sum_p A_c[(b,m), p] * w[o, c, p]
// with A_c sparse (4 nonzeros per row). K-chunks of 32 -> V_WMMA_F32_16X16X32_BF16.
// bf16 (not f16): clipped x (|x|>~2.1, ~1.7% of N(0,1) samples) yields basis
// magnitudes > 65504 (f16 overflow); bf16 keeps fp32 range, accumulate in f32.
//
// Parallelism note: the problem is tiny (134 MFLOP natural / ~4.4 MB traffic),
// so it is latency-bound at ~1-2 waves per SIMD. The two 16-wide o-tiles are
// split across separate waves (2048 waves total): half the WMMAs and half the
// B-loads on each wave's critical path, 2x the wave parallelism. Two
// accumulators (one per channel-parity plane) keep two independent XDL chains
// in flight.
// ---------------------------------------------------------------------------

typedef __attribute__((ext_vector_type(16))) __bf16 v16bf;
typedef __attribute__((ext_vector_type(8)))  __bf16 v8bf;
typedef __attribute__((ext_vector_type(8)))  float  v8f;

#define SEGS   32
#define INCH   32
#define OUTCH  32
#define BATCH  16
#define NODES  1024
#define WPTS   97      // (N-1)*SEGMENTS + 1
#define PPAD   128     // p axis padded to 4 K-chunks of 32

// ---------------------------------------------------------------------------
// Prep: w f32 [o][c][97]  ->  Bbuf bf16 in exact WMMA-B per-lane register image.
// Halfword index: c*4096 + q*1024 + otile*512 + lane*16 + kk
//   value = w[o = otile*16 + (lane&15)][c][p = 32q + 8*(lane>>4) + (kk&7) + 16*(kk>>3)], 0 if p>=97
// (16-bit B 32x16: lane half h holds K = 8h..8h+7 and K = 16+8h..16+8h+7)
// Total 131072 halfwords = 256 KB; L2-resident, shared by all tiles.
// ---------------------------------------------------------------------------
__global__ void prep_B_kernel(const float* __restrict__ w, __bf16* __restrict__ Bbuf) {
  int idx   = blockIdx.x * 256 + threadIdx.x;   // 131072 total
  int kk    = idx & 15;
  int lane  = (idx >> 4) & 31;
  int otile = (idx >> 9) & 1;
  int q     = (idx >> 10) & 3;
  int c     = (idx >> 12) & 31;
  int o     = otile * 16 + (lane & 15);
  int p     = 32 * q + 8 * (lane >> 4) + (kk & 7) + 16 * (kk >> 3);
  float v   = (p < WPTS) ? w[(o * INCH + c) * WPTS + p] : 0.0f;
  Bbuf[idx] = (__bf16)v;
}

__device__ __forceinline__ void eval_basis(float xv, int& s_out, float b[4]) {
  float u = (xv + 1.0f) * 16.0f;            // (x+HALF)/LENGTH*SEGMENTS
  int s = (int)u;                           // trunc-toward-zero == astype(int32)
  s = s < 0 ? 0 : (s > SEGS - 1 ? SEGS - 1 : s);
  float t = 32.0f * xv - 2.0f * (float)s + 31.0f;   // x_in in [-1,1] inside segment
  // Lagrange basis at Chebyshev nodes X = {-1, -0.5, 0.5, 1}
  float ta = t + 1.0f, tb = t + 0.5f, tc = t - 0.5f, td = t - 1.0f;
  b[0] = tb * tc * td * (-2.0f / 3.0f);
  b[1] = ta * tc * td * ( 4.0f / 3.0f);
  b[2] = ta * tb * td * (-4.0f / 3.0f);
  b[3] = ta * tb * tc * ( 2.0f / 3.0f);
  s_out = s;
}

// ---------------------------------------------------------------------------
// Main: one wave (32 lanes) per (16-point tile, o-tile) pair; 2048 waves.
// Lanes 0-15 build the A-plane for the even channel, 16-31 for the odd one;
// each lane exclusively owns one 128-entry bf16 row in LDS. Zero-old is
// unconditional (rows start zeroed, so clearing p=0..3 on iter 0 is a no-op),
// keeping the loop branch-free with EXEC all-ones throughout.
// 8 WMMAs per channel-pair -> 128 WMMAs per wave into two v8f accumulators.
// ---------------------------------------------------------------------------
__global__ void __launch_bounds__(32)
pw_wmma_kernel(const float* __restrict__ x, const __bf16* __restrict__ Bbuf,
               float* __restrict__ out) {
  __shared__ __align__(16) __bf16 Apl[2][16][PPAD];   // 8 KB

  const int lane  = threadIdx.x;
  const int h     = lane >> 4;          // plane (channel parity) / K-half for A
  const int r     = lane & 15;          // row (point) in tile / o within o-tile
  const int otile = blockIdx.x & 1;     // which 16 of the 32 output channels
  const int tile  = blockIdx.x >> 1;    // 1024 point-tiles
  const int b     = tile >> 6;
  const int m0    = (tile & 63) << 4;

  // Each lane zeroes the row it owns (256 B), once.
  {
    int4  z    = make_int4(0, 0, 0, 0);
    int4* rowp = (int4*)&Apl[h][r][0];
#pragma unroll
    for (int i = 0; i < 16; ++i) rowp[i] = z;
  }

  v8f acc0 = {};   // plane-0 (even channel) XDL chain
  v8f acc1 = {};   // plane-1 (odd channel) XDL chain
  int prev_s = 0;  // row starts zeroed -> clearing p=0..3 on iter 0 is harmless
  const float* xlane = x + (b * INCH + h) * NODES + m0 + r;  // +2*cp*NODES / iter

  for (int cp = 0; cp < 16; ++cp) {
    // -------- build sparse A rows for channels c = 2cp (plane 0) / 2cp+1 (plane 1)
    float xv = xlane[2 * cp * NODES];
    if (cp < 15) {
      // global_prefetch_b8, WGP scope: pull next iteration's x line into all
      // cache levels (latency-bound dispatch).
      __builtin_prefetch(&xlane[2 * (cp + 1) * NODES], 0, 3);
    }
    int s; float bs[4];
    eval_basis(xv, s, bs);

    __bf16* row = &Apl[h][r][0];
    int pp = 3 * prev_s;                 // clear last iteration's 4 entries
    row[pp] = (__bf16)0.0f; row[pp + 1] = (__bf16)0.0f;
    row[pp + 2] = (__bf16)0.0f; row[pp + 3] = (__bf16)0.0f;
    int p0 = 3 * s;
    row[p0]     = (__bf16)bs[0];
    row[p0 + 1] = (__bf16)bs[1];
    row[p0 + 2] = (__bf16)bs[2];
    row[p0 + 3] = (__bf16)bs[3];
    prev_s = s;

    // Same-wave LDS is processed in order; wait + compiler fence before reads.
    asm volatile("s_wait_dscnt 0" ::: "memory");

    // -------- 16x16x32 bf16 WMMAs over both planes, 4 K-chunks each
#pragma unroll
    for (int pl = 0; pl < 2; ++pl) {
      const int cc = 2 * cp + pl;
#pragma unroll
      for (int q = 0; q < 4; ++q) {
        // 16-bit A 16x32 layout: lane (r,h) holds K = 8h..8h+7 and K = 16+8h..
        v8bf alo = *(const v8bf*)&Apl[pl][r][q * 32 + h * 8];
        v8bf ahi = *(const v8bf*)&Apl[pl][r][q * 32 + h * 8 + 16];
        v16bf A = __builtin_shufflevector(alo, ahi,
            0, 1, 2, 3, 4, 5, 6, 7, 8, 9, 10, 11, 12, 13, 14, 15);

        const __bf16* bp = Bbuf + cc * 4096 + q * 1024 + otile * 512 + lane * 16;
        v16bf B = *(const v16bf*)bp;

        if (pl == 0)
          acc0 = __builtin_amdgcn_wmma_f32_16x16x32_bf16(
              false, A, false, B, (short)0, acc0, false, false);
        else
          acc1 = __builtin_amdgcn_wmma_f32_16x16x32_bf16(
              false, A, false, B, (short)0, acc1, false, false);
      }
    }
  }

  // Merge the two XDL chains.
  v8f acc = acc0 + acc1;

  // D layout: lane L, VGPR v -> D[M = v + 8*(L>>4), N = L&15]; M = point, N = o.
  // Lane's 8 values are m-consecutive -> two float4 stores.
  float* op = out + (b * OUTCH + otile * 16 + r) * NODES + m0 + 8 * h;
  *(float4*)(op)     = make_float4(acc[0], acc[1], acc[2], acc[3]);
  *(float4*)(op + 4) = make_float4(acc[4], acc[5], acc[6], acc[7]);
}

// ---------------------------------------------------------------------------
// Fallback (exact fp32, VALU gather) if workspace is too small for Bbuf.
// One wave per (b,m); lane = o; segment/basis uniform across lanes.
// ---------------------------------------------------------------------------
__global__ void __launch_bounds__(32)
pw_naive_kernel(const float* __restrict__ x, const float* __restrict__ w,
                float* __restrict__ out) {
  int bm = blockIdx.x;           // BATCH*NODES
  int b  = bm >> 10;
  int m  = bm & (NODES - 1);
  int o  = threadIdx.x;
  float acc = 0.0f;
  for (int c = 0; c < INCH; ++c) {
    float xv = x[(b * INCH + c) * NODES + m];
    int s; float bs[4];
    eval_basis(xv, s, bs);
    const float* wp = &w[(o * INCH + c) * WPTS + 3 * s];
    acc += bs[0] * wp[0] + bs[1] * wp[1] + bs[2] * wp[2] + bs[3] * wp[3];
  }
  out[(b * OUTCH + o) * NODES + m] = acc;
}

extern "C" void kernel_launch(void* const* d_in, const int* in_sizes, int n_in,
                              void* d_out, int out_size, void* d_ws, size_t ws_size,
                              hipStream_t stream) {
  const float* x = (const float*)d_in[0];   // (16, 32, 1024) f32
  const float* w = (const float*)d_in[1];   // (32, 32, 97)   f32
  float* out = (float*)d_out;               // (16, 32, 1024) f32

  const size_t bbuf_bytes = 131072 * sizeof(__bf16);   // 256 KB
  if (ws_size >= bbuf_bytes) {
    __bf16* Bbuf = (__bf16*)d_ws;
    prep_B_kernel<<<512, 256, 0, stream>>>(w, Bbuf);
    // 2048 waves: (1024 point-tiles) x (2 o-tiles)
    pw_wmma_kernel<<<(BATCH * NODES / 16) * 2, 32, 0, stream>>>(x, Bbuf, out);
  } else {
    pw_naive_kernel<<<BATCH * NODES, 32, 0, stream>>>(x, w, out);
  }
}
